// Encoder_9955734192583
// MI455X (gfx1250) — matrix operand order repaired and checked by reading, not compiled, not run on hardware
//
#include <hip/hip_runtime.h>

#define IN_DIM 512
#define HD1 128
#define HD2 64

typedef __bf16 bf16_t;
typedef bf16_t v16bf __attribute__((ext_vector_type(16)));
typedef float  v8f   __attribute__((ext_vector_type(8)));
typedef unsigned u32x4 __attribute__((ext_vector_type(4)));
typedef unsigned u32x8 __attribute__((ext_vector_type(8)));

__device__ __forceinline__ bf16_t f2bf(float f) {
  unsigned u = __builtin_bit_cast(unsigned, f);
  u += 0x7FFFu + ((u >> 16) & 1u);           // round-to-nearest-even
  unsigned short h = (unsigned short)(u >> 16);
  return __builtin_bit_cast(bf16_t, h);
}

__device__ __forceinline__ void atomAddF(float* p, float v) {
  __hip_atomic_fetch_add(p, v, __ATOMIC_RELAXED, __HIP_MEMORY_SCOPE_AGENT);
}

__global__ void zero_f32_kernel(float* __restrict__ p, int n) {
  int t = blockIdx.x * blockDim.x + threadIdx.x;
  if (t < n) p[t] = 0.0f;
}

__global__ void deg_count_kernel(const int* __restrict__ dst, int E, float* __restrict__ deg) {
  int t = blockIdx.x * blockDim.x + threadIdx.x;
  if (t < E) atomAddF(&deg[dst[t]], 1.0f);
}

__global__ void dinv_kernel(float* __restrict__ deg, int n) {
  int t = blockIdx.x * blockDim.x + threadIdx.x;
  if (t < n) deg[t] = rsqrtf(deg[t] + 1.0f);   // +1 for self-loop; always > 0
}

// One-time: Wt[col][k] = bf16(W[k][col])  (transposed, TDM/WMMA-friendly layout)
template<int NCOLS, int KDIM>
__global__ void wt_prep_kernel(const float* __restrict__ W, bf16_t* __restrict__ Wt) {
  int t = blockIdx.x * blockDim.x + threadIdx.x;
  if (t >= NCOLS * KDIM) return;
  int col = t / KDIM, k = t % KDIM;
  Wt[t] = f2bf(W[(size_t)k * NCOLS + col]);
}

// C[M x NCOLS] = A[M x KDIM] @ W[KDIM x NCOLS]; W pre-converted to bf16 as Wt[col][k].
// 4 waves/block; each wave owns one 16-row strip, all columns.
// Per K-step, wave 0 DMAs the 32 x NCOLS bf16 W-slice into LDS via the Tensor Data
// Mover (tensor_load_to_lds), synced with s_wait_tensorcnt + workgroup barriers.
template<int NCOLS, int KDIM>
__global__ __launch_bounds__(128)
void gemm_bf16_wmma_kernel(const float* __restrict__ A, const bf16_t* __restrict__ Wt,
                           float* __restrict__ C, int M)
{
  constexpr int NT = NCOLS / 16;
  __shared__ alignas(64) bf16_t Bt[NCOLS * 32];   // [col][k] bf16, 64B per col

  const int lane  = threadIdx.x & 31;
  const int wave  = threadIdx.x >> 5;
  const int strip = blockIdx.x * 4 + wave;
  const bool active = (strip * 16) < M;           // wave-uniform -> EXEC all-1s at WMMA
  const int half = lane >> 4;
  const int lrow = lane & 15;
  const size_t row = (size_t)strip * 16 + lrow;

  // --- TDM descriptor (ISA 8.3/8.4), loop-invariant part (group 1) ---
  const unsigned ldsOff =
      (unsigned)(unsigned long long)(__attribute__((address_space(3))) bf16_t*)&Bt[0];
  const unsigned long long gbase = (unsigned long long)(const void*)Wt;

  u32x8 g1;
  g1[0] = (1u << 16);                                   // data_size = 2 bytes
  g1[1] = ((unsigned)KDIM & 0xFFFFu) << 16;             // tensor_dim0[15:0]
  g1[2] = (((unsigned)KDIM >> 16) & 0xFFFFu)            // tensor_dim0[31:16]
        | (((unsigned)NCOLS & 0xFFFFu) << 16);          // tensor_dim1[15:0]
  g1[3] = (((unsigned)NCOLS >> 16) & 0xFFFFu)           // tensor_dim1[31:16]
        | (32u << 16);                                  // tile_dim0 = 32 elements
  g1[4] = (unsigned)NCOLS;                              // tile_dim1 = NCOLS, tile_dim2 = 0
  g1[5] = (unsigned)KDIM;                               // tensor_dim0_stride[31:0]
  g1[6] = 0;                                            // stride0 hi / stride1 lo
  g1[7] = 0;

  v8f acc[NT];
#pragma unroll
  for (int t = 0; t < NT; ++t) { v8f z = {}; acc[t] = z; }

  for (int kk = 0; kk < KDIM; kk += 32) {
    __syncthreads();                                    // prior-slice reads complete
    if (wave == 0) {
      const unsigned long long ga = gbase + (unsigned long long)kk * 2ull;
      u32x4 g0;
      g0[0] = 1u;                                       // count=1, user descriptor
      g0[1] = ldsOff;                                   // lds_addr (bytes)
      g0[2] = (unsigned)ga;                             // global_addr[31:0]
      g0[3] = ((unsigned)(ga >> 32) & 0x01FFFFFFu)      // global_addr[56:32]
            | 0x80000000u;                              // type = 2 ("image")
      asm volatile("tensor_load_to_lds %0, %1" :: "s"(g0), "s"(g1) : "memory");
      __builtin_amdgcn_s_wait_tensorcnt(0);
    }
    __syncthreads();                                    // LDS slice visible to all waves

    if (active) {
      // A fragment: lane row = lrow, K = half*16 + e
      const float4* ap = (const float4*)(A + row * KDIM + kk + half * 16);
      __builtin_prefetch((const float*)ap + 32, 0, 0);  // next K-slice
      const float4 a0 = ap[0], a1 = ap[1], a2 = ap[2], a3 = ap[3];
      v16bf afrag;
      afrag[0]  = f2bf(a0.x); afrag[1]  = f2bf(a0.y); afrag[2]  = f2bf(a0.z); afrag[3]  = f2bf(a0.w);
      afrag[4]  = f2bf(a1.x); afrag[5]  = f2bf(a1.y); afrag[6]  = f2bf(a1.z); afrag[7]  = f2bf(a1.w);
      afrag[8]  = f2bf(a2.x); afrag[9]  = f2bf(a2.y); afrag[10] = f2bf(a2.z); afrag[11] = f2bf(a2.w);
      afrag[12] = f2bf(a3.x); afrag[13] = f2bf(a3.y); afrag[14] = f2bf(a3.z); afrag[15] = f2bf(a3.w);

#pragma unroll
      for (int t = 0; t < NT; ++t) {
        // B fragment: col = t*16 + lrow, K = half*16 + e -> aligned 32B in Bt
        const v16bf bfrag = *(const v16bf*)&Bt[(t * 16 + lrow) * 32 + half * 16];
        acc[t] = __builtin_amdgcn_wmma_f32_16x16x32_bf16(
            false, afrag, false, bfrag, (short)0, acc[t], false, false);
      }
    }
  }

  if (active) {
#pragma unroll
    for (int t = 0; t < NT; ++t) {
#pragma unroll
      for (int r = 0; r < 8; ++r) {
        size_t m = (size_t)strip * 16 + r + half * 8;   // C layout: M = r + half*8
        C[m * NCOLS + t * 16 + lrow] = acc[t][r];
      }
    }
  }
}

// agg[dst] += dinv[src]*dinv[dst] * h[src], F/4 lanes per edge, float4 per lane.
template<int F>
__global__ void scatter_kernel(const int* __restrict__ src, const int* __restrict__ dst,
                               const float* __restrict__ dinv, const float* __restrict__ h,
                               float* agg, int E)
{
  constexpr int LPE = F / 4;
  int t = blockIdx.x * blockDim.x + threadIdx.x;
  int e = t / LPE;
  int l = t % LPE;
  if (e >= E) return;
  int s = src[e], d = dst[e];
  float nrm = dinv[s] * dinv[d];
  const float4 hv = *(const float4*)(h + (size_t)s * F + l * 4);
  float* ap = agg + (size_t)d * F + l * 4;
  atomAddF(ap + 0, nrm * hv.x);
  atomAddF(ap + 1, nrm * hv.y);
  atomAddF(ap + 2, nrm * hv.z);
  atomAddF(ap + 3, nrm * hv.w);
}

// out = relu(agg + dinv[i]^2 * h + bias)   (self-loop folded in, no atomics)
template<int F>
__global__ void finalize_kernel(const float* agg, const float* h,
                                const float* __restrict__ dinv, const float* __restrict__ bias,
                                float* out, int N)
{
  int t = blockIdx.x * blockDim.x + threadIdx.x;
  if (t >= N * F) return;
  int i = t / F, c = t % F;
  float di = dinv[i];
  float v = agg[t] + di * di * h[t] + bias[c];
  out[t] = v > 0.0f ? v : 0.0f;
}

static inline int cdiv_ll(long long a, long long b) { return (int)((a + b - 1) / b); }

extern "C" void kernel_launch(void* const* d_in, const int* in_sizes, int n_in,
                              void* d_out, int out_size, void* d_ws, size_t ws_size,
                              hipStream_t stream) {
  const float* x  = (const float*)d_in[0];
  const int*   ei = (const int*)d_in[1];   // [2, E] row-major: src then dst
  const float* W1 = (const float*)d_in[2];
  const float* b1 = (const float*)d_in[3];
  const float* W2 = (const float*)d_in[4];
  const float* b2 = (const float*)d_in[5];

  const int N = in_sizes[0] / IN_DIM;
  const int E = in_sizes[1] / 2;
  const int* srcIdx = ei;
  const int* dstIdx = ei + E;

  // workspace layout (256B-aligned given N=50000)
  float* deg  = (float*)d_ws;                        // N floats, becomes dinv in-place
  float* h1   = deg  + (((size_t)N + 63) & ~(size_t)63);
  float* agg1 = h1   + (size_t)N * HD1;              // becomes a1 (post-ReLU) in-place
  float* h2   = agg1 + (size_t)N * HD1;
  float* agg2 = h2   + (size_t)N * HD2;
  bf16_t* Wt1 = (bf16_t*)(agg2 + (size_t)N * HD2);   // [HD1][IN_DIM] bf16
  bf16_t* Wt2 = Wt1 + (size_t)HD1 * IN_DIM;          // [HD2][HD1]   bf16
  (void)n_in; (void)out_size; (void)ws_size;

  // zero accumulators (every call: atomics accumulate across graph replays)
  zero_f32_kernel<<<cdiv_ll(N, 256), 256, 0, stream>>>(deg, N);
  zero_f32_kernel<<<cdiv_ll((long long)N * HD1, 256), 256, 0, stream>>>(agg1, N * HD1);
  zero_f32_kernel<<<cdiv_ll((long long)N * HD2, 256), 256, 0, stream>>>(agg2, N * HD2);

  deg_count_kernel<<<cdiv_ll(E, 256), 256, 0, stream>>>(dstIdx, E, deg);
  dinv_kernel<<<cdiv_ll(N, 256), 256, 0, stream>>>(deg, N);

  wt_prep_kernel<HD1, IN_DIM><<<cdiv_ll(HD1 * IN_DIM, 256), 256, 0, stream>>>(W1, Wt1);
  wt_prep_kernel<HD2, HD1><<<cdiv_ll(HD2 * HD1, 256), 256, 0, stream>>>(W2, Wt2);

  const int strips = (N + 15) / 16;
  const int gemmBlocks = (strips + 3) / 4;

  // Layer 1
  gemm_bf16_wmma_kernel<HD1, IN_DIM><<<gemmBlocks, 128, 0, stream>>>(x, Wt1, h1, N);
  scatter_kernel<HD1><<<cdiv_ll((long long)E * (HD1 / 4), 256), 256, 0, stream>>>(
      srcIdx, dstIdx, deg, h1, agg1, E);
  finalize_kernel<HD1><<<cdiv_ll((long long)N * HD1, 256), 256, 0, stream>>>(
      agg1, h1, deg, b1, agg1, N);

  // Layer 2
  gemm_bf16_wmma_kernel<HD2, HD1><<<gemmBlocks, 128, 0, stream>>>(agg1, Wt2, h2, N);
  scatter_kernel<HD2><<<cdiv_ll((long long)E * (HD2 / 4), 256), 256, 0, stream>>>(
      srcIdx, dstIdx, deg, h2, agg2, E);
  finalize_kernel<HD2><<<cdiv_ll((long long)N * HD2, 256), 256, 0, stream>>>(
      agg2, h2, deg, b2, (float*)d_out, N);
}
